// SelfAttention_34617436405978
// MI455X (gfx1250) — compile-verified, matrix operand order checked by reading, not run on hardware
//
#include <hip/hip_runtime.h>

#define NBATCH 2
#define SEQ    1024
#define EMB    1024
#define NH     16
#define HD     64

typedef __attribute__((ext_vector_type(16))) __bf16 v16bf;
typedef __attribute__((ext_vector_type(8)))  float  v8f;

union FragU { uint4 q[2]; v16bf v; };

static __device__ __forceinline__ unsigned short f2bf(float f) {
  union { float f; unsigned u; } x; x.f = f;
  unsigned u = x.u;
  u += 0x7FFFu + ((u >> 16) & 1u);   // round-to-nearest-even
  return (unsigned short)(u >> 16);
}

// Load one 16x32 bf16 WMMA A/B fragment from a row-major [row][col] bf16
// (ushort) array. Per wave32 ISA layout: lane holds row (lane&15); two
// contiguous 8-element K runs at colBase + (lane>>4)*8 and +16 more.
static __device__ __forceinline__ v16bf load_frag(const unsigned short* base,
                                                  int rowBase, int colBase,
                                                  int stride) {
  int lane = threadIdx.x & 31;
  int r = rowBase + (lane & 15);
  int c = colBase + ((lane >> 4) << 3);
  FragU f;
  f.q[0] = *(const uint4*)(base + r * stride + c);
  f.q[1] = *(const uint4*)(base + r * stride + c + 16);
  return f.v;
}

// One async 16-byte global->LDS copy (per lane). Tracked by ASYNCcnt.
static __device__ __forceinline__ void async_b128(const void* gsrc, void* ldst) {
  unsigned           lds = (unsigned)(size_t)ldst;          // as(3) offset
  unsigned long long ga  = (unsigned long long)(size_t)gsrc;
  asm volatile("global_load_async_to_lds_b128 %0, %1, off"
               :: "v"(lds), "v"(ga) : "memory");
}

#define WAIT_ASYNC(n) asm volatile("s_wait_asynccnt %0" :: "n"(n) : "memory")

// ---------------------------------------------------------------------------
// M[i][j] = (1/8) * sum_e Wq[e][i] * Wk[e][j]        (64x64 f32)
__global__ void k_prepM(const float* __restrict__ Wq, const float* __restrict__ Wk,
                        float* __restrict__ M) {
  int t = blockIdx.x * blockDim.x + threadIdx.x;     // 4096 threads
  int i = t >> 6, j = t & 63;
  float s = 0.f;
  for (int e = 0; e < EMB; ++e)
    s += Wq[e * HD + i] * Wk[e * HD + j];
  M[t] = s * 0.125f;                                 // fold 1/sqrt(hd)
}

// W2[o][d] = sum_e Wo[o][e] * Wv[e][d]               (1024x64 bf16 row-major)
__global__ void k_prepW2(const float* __restrict__ Wo, const float* __restrict__ Wv,
                         unsigned short* __restrict__ W2) {
  int t = blockIdx.x * blockDim.x + threadIdx.x;     // 65536 threads
  int o = t >> 6, d = t & 63;
  float s = 0.f;
  for (int e = 0; e < EMB; ++e)
    s += Wo[o * EMB + e] * Wv[e * HD + d];
  W2[t] = f2bf(s);
}

// qt[n][h][s][d] = bf16( sum_dp q_in[n][s][h][dp] * M[dp][d] )
__global__ void k_projQ(const float* __restrict__ Qin, const float* __restrict__ M,
                        unsigned short* __restrict__ qt) {
  int t = blockIdx.x * blockDim.x + threadIdx.x;     // 2*16*1024*64
  int d = t & 63;
  int s = (t >> 6) & (SEQ - 1);
  int h = (t >> 16) & (NH - 1);
  int n = t >> 20;
  const float* qr = Qin + (((n * SEQ + s) * NH + h) * HD);
  float acc = 0.f;
  for (int dp = 0; dp < HD; ++dp)
    acc += qr[dp] * M[dp * HD + d];
  qt[t] = f2bf(acc);
}

// kt[n][h][s][d] = bf16(k_in[n][s][h][d])   (head-major relayout)
__global__ void k_relayK(const float* __restrict__ Kin, unsigned short* __restrict__ kt) {
  int t = blockIdx.x * blockDim.x + threadIdx.x;
  int d = t & 63;
  int s = (t >> 6) & (SEQ - 1);
  int h = (t >> 16) & (NH - 1);
  int n = t >> 20;
  kt[t] = f2bf(Kin[((n * SEQ + s) * NH + h) * HD + d]);
}

// vtT[n][h][d][s] = bf16(v_in[n][s][h][d])  (head-major + transpose)
__global__ void k_relayV(const float* __restrict__ Vin, unsigned short* __restrict__ vt) {
  int t = blockIdx.x * blockDim.x + threadIdx.x;
  int s = t & (SEQ - 1);
  int d = (t >> 10) & 63;
  int h = (t >> 16) & (NH - 1);
  int n = t >> 20;
  vt[t] = f2bf(Vin[((n * SEQ + s) * NH + h) * HD + d]);
}

// ---------------------------------------------------------------------------
// Flash attention over one head: 4 waves/block, each wave owns 16 q rows.
// K/V tiles double-buffered in LDS via async global->LDS DMA (ASYNCcnt).
__launch_bounds__(128, 1)
__global__ void k_attn(const unsigned short* __restrict__ qt,
                       const unsigned short* __restrict__ kt,
                       const unsigned short* __restrict__ vt,
                       float* __restrict__ ov) {
  __shared__ __align__(16) unsigned short Ks[2][64 * 64];   // [buf][k_row][d]
  __shared__ __align__(16) unsigned short Vs[2][64 * 64];   // [buf][d][k_row]
  __shared__ __align__(16) unsigned short Ps[4][16 * 64];   // per-wave P

  const int tid  = threadIdx.x;
  const int lane = tid & 31;
  const int w    = tid >> 5;
  const int half = lane >> 4;

  const int qt64 = blockIdx.x & 15;
  const int h    = (blockIdx.x >> 4) & 15;
  const int n    = blockIdx.x >> 8;

  const unsigned short* qhead = qt + (size_t)((n * NH + h) * SEQ) * HD;
  const unsigned short* khead = kt + (size_t)((n * NH + h) * SEQ) * HD;
  const unsigned short* vhead = vt + (size_t)((n * NH + h) * HD) * SEQ;

  const int qRow0 = qt64 * 64 + w * 16;

  v16bf aq0 = load_frag(qhead + qRow0 * HD, 0, 0, HD);
  v16bf aq1 = load_frag(qhead + qRow0 * HD, 0, 32, HD);

  v8f o[4] = {};
  float m_run[8], l_run[8];
#pragma unroll
  for (int r = 0; r < 8; ++r) { m_run[r] = -3.0e38f; l_run[r] = 0.f; }

  // Issue one K/V tile (8 async b128 ops per thread-batch: 4 for K, 4 for V).
  auto issue_tile = [&](int buf, int kc) {
    const unsigned short* ksrc = khead + kc * 64 * HD;   // contiguous 8KB
#pragma unroll
    for (int i = 0; i < 4; ++i) {
      int u = tid + i * 128;                             // 512 x uint4
      async_b128(ksrc + u * 8, &Ks[buf][u * 8]);
    }
#pragma unroll
    for (int i = 0; i < 4; ++i) {
      int u = tid + i * 128;                             // 8 uint4 per d-row
      int d = u >> 3, cu = u & 7;
      async_b128(vhead + d * SEQ + kc * 64 + cu * 8, &Vs[buf][u * 8]);
    }
  };

  issue_tile(0, 0);

  constexpr int NT = SEQ / 64;
  for (int kc = 0; kc < NT; ++kc) {
    const int cur = kc & 1;
    __syncthreads();                  // all waves done reading buf 1-cur
    if (kc + 1 < NT) {
      issue_tile(1 - cur, kc + 1);    // prefetch next tile
      WAIT_ASYNC(8);                  // in-order: batch kc complete
    } else {
      WAIT_ASYNC(0);
    }
    __syncthreads();                  // tile `cur` visible to all waves

    const unsigned short* KsC = Ks[cur];
    const unsigned short* VsC = Vs[cur];

    // scores tile 16x64 = 4 column blocks, K-dim = hd = 64 (2 WMMA steps)
    v8f c[4];
#pragma unroll
    for (int nb = 0; nb < 4; ++nb) {
      v16bf b0 = load_frag(KsC, nb * 16, 0, HD);
      v16bf b1 = load_frag(KsC, nb * 16, 32, HD);
      v8f acc = {};
      acc = __builtin_amdgcn_wmma_f32_16x16x32_bf16(false, aq0, false, b0,
                                                    (short)0, acc, false, false);
      acc = __builtin_amdgcn_wmma_f32_16x16x32_bf16(false, aq1, false, b1,
                                                    (short)0, acc, false, false);
      c[nb] = acc;
    }

    // online softmax; row r lives in VGPR r across a 16-lane half
    unsigned short* PsW = &Ps[w][0];
#pragma unroll
    for (int r = 0; r < 8; ++r) {
      float mn = fmaxf(fmaxf(c[0][r], c[1][r]), fmaxf(c[2][r], c[3][r]));
#pragma unroll
      for (int off = 1; off < 16; off <<= 1)
        mn = fmaxf(mn, __shfl_xor(mn, off, 32));
      float mnew = fmaxf(m_run[r], mn);
      float alpha = __expf(m_run[r] - mnew);
      m_run[r] = mnew;
      float p0 = __expf(c[0][r] - mnew);
      float p1 = __expf(c[1][r] - mnew);
      float p2 = __expf(c[2][r] - mnew);
      float p3 = __expf(c[3][r] - mnew);
      float rs = p0 + p1 + p2 + p3;
#pragma unroll
      for (int off = 1; off < 16; off <<= 1)
        rs += __shfl_xor(rs, off, 32);
      l_run[r] = l_run[r] * alpha + rs;
#pragma unroll
      for (int i = 0; i < 4; ++i) o[i][r] *= alpha;
      int row = r + (half << 3);
      int col = lane & 15;
      PsW[row * 64 +  0 + col] = f2bf(p0);
      PsW[row * 64 + 16 + col] = f2bf(p1);
      PsW[row * 64 + 32 + col] = f2bf(p2);
      PsW[row * 64 + 48 + col] = f2bf(p3);
    }
    __syncthreads();   // order P LDS stores vs fragment reloads

    v16bf ap0 = load_frag(PsW, 0, 0, 64);
    v16bf ap1 = load_frag(PsW, 0, 32, 64);
#pragma unroll
    for (int db = 0; db < 4; ++db) {
      v16bf bv0 = load_frag(VsC, db * 16, 0, 64);
      v16bf bv1 = load_frag(VsC, db * 16, 32, 64);
      o[db] = __builtin_amdgcn_wmma_f32_16x16x32_bf16(false, ap0, false, bv0,
                                                      (short)0, o[db], false, false);
      o[db] = __builtin_amdgcn_wmma_f32_16x16x32_bf16(false, ap1, false, bv1,
                                                      (short)0, o[db], false, false);
    }
  }

  float* ovp = ov + ((size_t)((n * NH + h) * SEQ) + qRow0) * HD;
#pragma unroll
  for (int r = 0; r < 8; ++r) {
    float inv = 1.0f / l_run[r];
    int row = r + (half << 3);
    int col = lane & 15;
#pragma unroll
    for (int db = 0; db < 4; ++db)
      ovp[row * HD + db * 16 + col] = o[db][r] * inv;
  }
}

// ---------------------------------------------------------------------------
// out[n][s][o] = (mean_h ov[n][h][s][:]) · W2[o][:] + bo[o]
__global__ void k_out(const float* __restrict__ ov,
                      const unsigned short* __restrict__ W2,
                      const float* __restrict__ bo,
                      float* __restrict__ out) {
  __shared__ __align__(16) unsigned short As[16 * 64];
  const int lane  = threadIdx.x;     // one wave per block
  const int half  = lane >> 4;
  const int n     = blockIdx.x >> 6;
  const int sBase = (blockIdx.x & 63) * 16;

  for (int e = lane; e < 16 * 64; e += 32) {
    int row = e >> 6, d = e & 63;
    float s = 0.f;
#pragma unroll
    for (int hh = 0; hh < NH; ++hh)
      s += ov[((size_t)((n * NH + hh) * SEQ) + sBase + row) * HD + d];
    As[e] = f2bf(s * (1.0f / NH));
  }
  __syncthreads();

  v16bf a0 = load_frag(As, 0, 0, 64);
  v16bf a1 = load_frag(As, 0, 32, 64);

  for (int ob = 0; ob < EMB / 16; ++ob) {
    __builtin_prefetch(W2 + (ob + 1) * 16 * HD, 0, 0);
    v16bf b0 = load_frag(W2, ob * 16, 0, HD);
    v16bf b1 = load_frag(W2, ob * 16, 32, HD);
    v8f c = {};
    c = __builtin_amdgcn_wmma_f32_16x16x32_bf16(false, a0, false, b0,
                                                (short)0, c, false, false);
    c = __builtin_amdgcn_wmma_f32_16x16x32_bf16(false, a1, false, b1,
                                                (short)0, c, false, false);
    int col = ob * 16 + (lane & 15);
    float bias = bo[col];
#pragma unroll
    for (int r = 0; r < 8; ++r) {
      int row = sBase + r + (half << 3);
      out[(size_t)(n * SEQ + row) * EMB + col] = c[r] + bias;
    }
  }
}

// ---------------------------------------------------------------------------
extern "C" void kernel_launch(void* const* d_in, const int* in_sizes, int n_in,
                              void* d_out, int out_size, void* d_ws, size_t ws_size,
                              hipStream_t stream) {
  (void)in_sizes; (void)n_in; (void)out_size; (void)ws_size;
  const float* Kin = (const float*)d_in[0];
  const float* Qin = (const float*)d_in[1];
  const float* Vin = (const float*)d_in[2];
  const float* Wk  = (const float*)d_in[3];
  const float* Wq  = (const float*)d_in[4];
  const float* Wv  = (const float*)d_in[5];
  const float* Wo  = (const float*)d_in[6];
  const float* bo  = (const float*)d_in[7];
  float* out = (float*)d_out;

  char* ws = (char*)d_ws;
  float*          Mm = (float*)(ws + 0);                 //  16 KB
  unsigned short* W2 = (unsigned short*)(ws + 16384);    // 128 KB
  unsigned short* qt = (unsigned short*)(ws + 147456);   //   4 MB
  unsigned short* kt = (unsigned short*)(ws + 4341760);  //   4 MB
  unsigned short* vt = (unsigned short*)(ws + 8536064);  //   4 MB
  float*          ov = (float*)(ws + 12730368);          //   8 MB

  k_prepM <<<16,   256, 0, stream>>>(Wq, Wk, Mm);
  k_prepW2<<<256,  256, 0, stream>>>(Wo, Wv, W2);
  k_projQ <<<8192, 256, 0, stream>>>(Qin, Mm, qt);
  k_relayK<<<8192, 256, 0, stream>>>(Kin, kt);
  k_relayV<<<8192, 256, 0, stream>>>(Vin, vt);
  k_attn  <<<512,  128, 0, stream>>>(qt, kt, vt, ov);
  k_out   <<<128,  32,  0, stream>>>(ov, W2, bo, out);
}
